// Net_18064632447522
// MI455X (gfx1250) — compile-verified, matrix operand order checked by reading, not run on hardware
//
#include <hip/hip_runtime.h>
#include <hip/hip_bf16.h>

// ---------------------------------------------------------------------------
// Mamba stack for MI455X (gfx1250, wave32).
// GEMMs: bf16 V_WMMA_F32_16X16X32_BF16, 2x2 register tile per wave,
//        double-buffered global_load_b128 fragment loads.
// Scan:  fp32, 16 lanes = 16 states, 2 channels per wave32, fused gating.
// ---------------------------------------------------------------------------

typedef __attribute__((ext_vector_type(16))) __bf16 v16bf;
typedef __attribute__((ext_vector_type(8)))  float  v8f;

#define R_TOT  8192   // B*L
#define LSEQ   2048
#define NB     4
#define HIDN   512
#define DINX   1024
#define D2X    2048
#define NST    16     // DSTATE
#define DTRX   32
#define NTOTX  64     // DTR + 2*DSTATE

// ---------------- fp32 -> bf16 convert ----------------
__global__ void k_cvt_bf16(const float* __restrict__ s, __bf16* __restrict__ d, int n) {
  int i = blockIdx.x * blockDim.x + threadIdx.x;
  if (i < n) d[i] = (__bf16)s[i];
}

// ---------------- fragment loaders (CDNA5 16-bit A/B layout) ----------------
// lane&15 selects the row (A) / output column's weight row (B);
// lane>>4 selects which 8-wide K sub-block the lane's halves hold.
__device__ __forceinline__ v16bf ld_frag(const __bf16* __restrict__ base,
                                         int r, int ld, int koff) {
  const __bf16* p = base + (size_t)r * ld + koff;
  union { uint4 u[2]; v16bf v; } u;
  u.u[0] = *(const uint4*)(p);
  u.u[1] = *(const uint4*)(p + 16);
  return u.v;
}

__device__ __forceinline__ v16bf ld_frag_f32(const float* __restrict__ base,
                                             int r, int ld, int koff) {
  const float* p = base + (size_t)r * ld + koff;
  v16bf a;
  #pragma unroll
  for (int j = 0; j < 8; ++j) { a[j] = (__bf16)p[j]; a[8 + j] = (__bf16)p[16 + j]; }
  return a;
}

__device__ __forceinline__ v8f wmma_bf16(v16bf a, v16bf b, v8f c) {
  return __builtin_amdgcn_wmma_f32_16x16x32_bf16(false, a, false, b,
                                                 (short)0, c, false, false);
}

// EPI: 0 = store, 1 = +bias store, 2 = softplus(x+bias) store, 3 = C += acc
template<int EPI>
__device__ __forceinline__ void store_tile(v8f c, float* __restrict__ C,
                                           const float* __restrict__ bias,
                                           int m, int n, int ldc, int lane) {
  const int col = lane & 15;
  const int rb  = m + ((lane >> 4) << 3);
  float bv = 0.f;
  if constexpr (EPI == 1 || EPI == 2) bv = bias[n + col];
  #pragma unroll
  for (int v = 0; v < 8; ++v) {
    size_t idx = (size_t)(rb + v) * ldc + n + col;
    float x = c[v];
    if constexpr (EPI == 1) { x += bv; }
    if constexpr (EPI == 2) { x += bv; x = (x > 20.f) ? x : log1pf(__expf(x)); }
    if constexpr (EPI == 3) { x += C[idx]; }
    C[idx] = x;
  }
}

// ---------------- WMMA GEMM: C[M,N] = A[M,K] * W[N,K]^T ---------------------
// Block = 128 threads (4 waves) tiling 64x64; each wave owns a 32x32 tile
// as 2x2 accumulators (2 A frags x 2 B frags -> 4 WMMA per K-step).
// K fragments are double-buffered so loads overlap the matrix pipe.
template<int EPI, bool AFP32>
__global__ void k_gemm(const void* __restrict__ Araw, const __bf16* __restrict__ W,
                       float* __restrict__ C, const float* __restrict__ bias,
                       int K, int lda, int ldw, int ldc) {
  const int lane = threadIdx.x & 31;
  const int wave = threadIdx.x >> 5;
  const int mb = blockIdx.y * 64 + (wave >> 1) * 32;
  const int nb = blockIdx.x * 64 + (wave & 1) * 32;
  const int row = lane & 15;
  const int kh  = (lane >> 4) * 8;

  const int mr0 = mb + row,      mr1 = mb + 16 + row;
  const int nr0 = nb + row,      nr1 = nb + 16 + row;
  const __bf16* A16 = (const __bf16*)Araw;
  const float*  A32 = (const float*)Araw;

  v8f c00 = {}, c01 = {}, c10 = {}, c11 = {};
  v16bf a0, a1, b0, b1;
  if constexpr (AFP32) { a0 = ld_frag_f32(A32, mr0, lda, kh);
                         a1 = ld_frag_f32(A32, mr1, lda, kh); }
  else                 { a0 = ld_frag(A16, mr0, lda, kh);
                         a1 = ld_frag(A16, mr1, lda, kh); }
  b0 = ld_frag(W, nr0, ldw, kh);
  b1 = ld_frag(W, nr1, ldw, kh);

  for (int k0 = 32; k0 < K; k0 += 32) {
    v16bf na0, na1, nb0, nb1;
    if constexpr (AFP32) { na0 = ld_frag_f32(A32, mr0, lda, k0 + kh);
                           na1 = ld_frag_f32(A32, mr1, lda, k0 + kh); }
    else                 { na0 = ld_frag(A16, mr0, lda, k0 + kh);
                           na1 = ld_frag(A16, mr1, lda, k0 + kh); }
    nb0 = ld_frag(W, nr0, ldw, k0 + kh);
    nb1 = ld_frag(W, nr1, ldw, k0 + kh);
    c00 = wmma_bf16(a0, b0, c00);
    c01 = wmma_bf16(a0, b1, c01);
    c10 = wmma_bf16(a1, b0, c10);
    c11 = wmma_bf16(a1, b1, c11);
    a0 = na0; a1 = na1; b0 = nb0; b1 = nb1;
  }
  c00 = wmma_bf16(a0, b0, c00);
  c01 = wmma_bf16(a0, b1, c01);
  c10 = wmma_bf16(a1, b0, c10);
  c11 = wmma_bf16(a1, b1, c11);

  store_tile<EPI>(c00, C, bias, mb,      nb,      ldc, lane);
  store_tile<EPI>(c01, C, bias, mb,      nb + 16, ldc, lane);
  store_tile<EPI>(c10, C, bias, mb + 16, nb,      ldc, lane);
  store_tile<EPI>(c11, C, bias, mb + 16, nb + 16, ldc, lane);
}

// ---------------- RMSNorm: one wave per row, shfl reduction -----------------
__global__ void k_rmsnorm(const float* __restrict__ h, const float* __restrict__ w,
                          __bf16* __restrict__ o) {
  const int lane = threadIdx.x & 31;
  const int r = blockIdx.x * 8 + (threadIdx.x >> 5);
  const float* hr = h + (size_t)r * HIDN;
  float vals[16];
  float ss = 0.f;
  #pragma unroll
  for (int j = 0; j < 16; ++j) { vals[j] = hr[lane + 32 * j]; ss += vals[j] * vals[j]; }
  for (int off = 16; off; off >>= 1) ss += __shfl_xor(ss, off, 32);
  const float scale = rsqrtf(ss * (1.f / HIDN) + 1e-5f);
  __bf16* orow = o + (size_t)r * HIDN;
  #pragma unroll
  for (int j = 0; j < 16; ++j)
    orow[lane + 32 * j] = (__bf16)(vals[j] * scale * w[lane + 32 * j]);
}

// ---------------- causal depthwise conv (DCONV=4) + SiLU --------------------
__global__ void k_conv_silu(const float* __restrict__ xz, const float* __restrict__ cw,
                            const float* __restrict__ cb, float* __restrict__ xa32,
                            __bf16* __restrict__ xa16) {
  int i = blockIdx.x * blockDim.x + threadIdx.x;    // over R_TOT*DINX
  int d = i & (DINX - 1);
  int r = i >> 10;
  int t = r & (LSEQ - 1);
  int b = r >> 11;
  const float* wr = cw + d * 4;
  float acc = cb[d];
  #pragma unroll
  for (int j = 0; j < 4; ++j) {
    int tt = t - 3 + j;
    if (tt >= 0) acc += xz[(size_t)((b << 11) + tt) * D2X + d] * wr[j];
  }
  float s = acc * (1.f / (1.f + __expf(-acc)));     // SiLU
  xa32[i] = s;
  xa16[i] = (__bf16)s;
}

// ---- selective scan + fused gate: y16 = (scan_y + D*x) * silu(z) -----------
// 16 lanes = 16 states of one (b,d) channel; 2 channels per wave32.
// Loads for step t+1 are issued before the step-t recurrence math so the
// dependent exp/fma/shuffle chain overlaps memory latency.
__global__ void k_scan_gate(const float* __restrict__ delta,
                            const float* __restrict__ xa32,
                            const float* __restrict__ dbc,
                            const float* __restrict__ A_log,
                            const float* __restrict__ xz,
                            const float* __restrict__ Dp,
                            __bf16* __restrict__ y16) {
  const int lane = threadIdx.x & 31;
  const int wave = blockIdx.x * (blockDim.x >> 5) + (threadIdx.x >> 5);
  const int n    = lane & 15;
  const int chan = wave * 2 + (lane >> 4);          // 0..4095 = b*DINX+d
  const int d    = chan & (DINX - 1);
  const int b    = chan >> 10;
  const float a  = -__expf(A_log[(size_t)d * NST + n]);
  const float Dd = Dp[d];
  const size_t rbase = (size_t)b * LSEQ;

  float h = 0.f;
  float dt = delta[rbase * DINX + d];
  float x  = xa32[rbase * DINX + d];
  float Bn = dbc[rbase * NTOTX + DTRX + n];
  float Cn = dbc[rbase * NTOTX + DTRX + NST + n];
  float z  = xz[rbase * D2X + DINX + d];

  for (int t = 0; t < LSEQ - 1; ++t) {
    const size_t rn = rbase + t + 1;
    float ndt = delta[rn * DINX + d];
    float nx  = xa32[rn * DINX + d];
    float nB  = dbc[rn * NTOTX + DTRX + n];
    float nC  = dbc[rn * NTOTX + DTRX + NST + n];
    float nz  = xz[rn * D2X + DINX + d];

    h = __expf(dt * a) * h + (dt * x) * Bn;
    float v = h * Cn;
    v += __shfl_xor(v, 1, 16);
    v += __shfl_xor(v, 2, 16);
    v += __shfl_xor(v, 4, 16);
    v += __shfl_xor(v, 8, 16);
    if (n == 0) {
      float sz = z * (1.f / (1.f + __expf(-z)));
      y16[(rbase + t) * DINX + d] = (__bf16)((v + Dd * x) * sz);
    }
    dt = ndt; x = nx; Bn = nB; Cn = nC; z = nz;
  }
  // tail step
  h = __expf(dt * a) * h + (dt * x) * Bn;
  float v = h * Cn;
  v += __shfl_xor(v, 1, 16);
  v += __shfl_xor(v, 2, 16);
  v += __shfl_xor(v, 4, 16);
  v += __shfl_xor(v, 8, 16);
  if (n == 0) {
    float sz = z * (1.f / (1.f + __expf(-z)));
    y16[(rbase + LSEQ - 1) * DINX + d] = (__bf16)((v + Dd * x) * sz);
  }
}

// ---------------- head: out = tanh(h . w_out + b), one wave per row ---------
__global__ void k_head(const float* __restrict__ h, const float* __restrict__ w,
                       const float* __restrict__ b, float* __restrict__ out) {
  const int lane = threadIdx.x & 31;
  const int r = blockIdx.x * 8 + (threadIdx.x >> 5);
  const float* hr = h + (size_t)r * HIDN;
  float s = 0.f;
  #pragma unroll
  for (int j = lane; j < HIDN; j += 32) s += hr[j] * w[j];
  for (int off = 16; off; off >>= 1) s += __shfl_xor(s, off, 32);
  if (lane == 0) out[r] = tanhf(s + b[0]);
}

// ---------------------------------------------------------------------------
extern "C" void kernel_launch(void* const* d_in, const int* in_sizes, int n_in,
                              void* d_out, int out_size, void* d_ws, size_t ws_size,
                              hipStream_t stream) {
  const float* x        = (const float*)d_in[0];
  const float* w_in     = (const float*)d_in[1];
  const float* b_in     = (const float*)d_in[2];
  const float* norm_w   = (const float*)d_in[3];
  const float* in_proj  = (const float*)d_in[4];
  const float* conv_w   = (const float*)d_in[5];
  const float* conv_b   = (const float*)d_in[6];
  const float* xproj    = (const float*)d_in[7];
  const float* dtproj   = (const float*)d_in[8];
  const float* dtproj_b = (const float*)d_in[9];
  const float* A_log    = (const float*)d_in[10];
  const float* D_param  = (const float*)d_in[11];
  const float* outproj  = (const float*)d_in[12];
  const float* w_out    = (const float*)d_in[13];
  const float* b_out    = (const float*)d_in[14];
  float* out = (float*)d_out;

  const int R = R_TOT;
  char* ws = (char*)d_ws;
  size_t off = 0;
  auto carve = [&](size_t bytes) -> char* {
    char* p = ws + off;
    off += (bytes + 255) & ~(size_t)255;
    return p;
  };

  __bf16* x16    = (__bf16*)carve((size_t)R * 32 * 2);
  __bf16* w_in16 = (__bf16*)carve((size_t)HIDN * 32 * 2);
  __bf16* inpw16 = (__bf16*)carve((size_t)NB * D2X * HIDN * 2);
  __bf16* xpw16  = (__bf16*)carve((size_t)NB * NTOTX * DINX * 2);
  __bf16* dtw16  = (__bf16*)carve((size_t)NB * DINX * DTRX * 2);
  __bf16* outw16 = (__bf16*)carve((size_t)NB * HIDN * DINX * 2);
  float*  h      = (float*) carve((size_t)R * HIDN * 4);
  __bf16* hn16   = (__bf16*)carve((size_t)R * HIDN * 2);
  float*  xz     = (float*) carve((size_t)R * D2X * 4);
  float*  xa32   = (float*) carve((size_t)R * DINX * 4);
  __bf16* xa16   = (__bf16*)carve((size_t)R * DINX * 2);
  float*  dbc    = (float*) carve((size_t)R * NTOTX * 4);
  float*  delta  = (float*) carve((size_t)R * DINX * 4);

  auto cvt = [&](const float* s, __bf16* d, int n) {
    k_cvt_bf16<<<(n + 255) / 256, 256, 0, stream>>>(s, d, n);
  };
  cvt(x,       x16,    R * 32);
  cvt(w_in,    w_in16, HIDN * 32);
  cvt(in_proj, inpw16, NB * D2X * HIDN);
  cvt(xproj,   xpw16,  NB * NTOTX * DINX);
  cvt(dtproj,  dtw16,  NB * DINX * DTRX);
  cvt(outproj, outw16, NB * HIDN * DINX);

  // h = x @ w_in^T + b_in  (K=32 -> single WMMA step per tile)
  k_gemm<1, false><<<dim3(HIDN / 64, R / 64), 128, 0, stream>>>(
      x16, w_in16, h, b_in, 32, 32, 32, HIDN);

  for (int i = 0; i < NB; ++i) {
    k_rmsnorm<<<R / 8, 256, 0, stream>>>(h, norm_w + (size_t)i * HIDN, hn16);

    k_gemm<0, false><<<dim3(D2X / 64, R / 64), 128, 0, stream>>>(
        hn16, inpw16 + (size_t)i * D2X * HIDN, xz, nullptr, HIDN, HIDN, HIDN, D2X);

    k_conv_silu<<<(R * DINX) / 256, 256, 0, stream>>>(
        xz, conv_w + (size_t)i * DINX * 4, conv_b + (size_t)i * DINX, xa32, xa16);

    k_gemm<0, false><<<dim3(NTOTX / 64, R / 64), 128, 0, stream>>>(
        xa16, xpw16 + (size_t)i * NTOTX * DINX, dbc, nullptr, DINX, DINX, DINX, NTOTX);

    // delta = softplus(dt @ dt_w^T + dt_b); A is fp32 dbc (first 32 cols)
    k_gemm<2, true><<<dim3(DINX / 64, R / 64), 128, 0, stream>>>(
        dbc, dtw16 + (size_t)i * DINX * DTRX, delta, dtproj_b + (size_t)i * DINX,
        32, NTOTX, DTRX, DINX);

    // scan + gate fused: writes bf16 y directly into xa16
    k_scan_gate<<<256, 256, 0, stream>>>(
        delta, xa32, dbc, A_log + (size_t)i * DINX * NST, xz,
        D_param + (size_t)i * DINX, xa16);

    // h += y @ out_w^T  (fused residual)
    k_gemm<3, false><<<dim3(HIDN / 64, R / 64), 128, 0, stream>>>(
        xa16, outw16 + (size_t)i * HIDN * DINX, h, nullptr, DINX, DINX, DINX, HIDN);
  }

  k_head<<<R / 8, 256, 0, stream>>>(h, w_out, b_out, out);
}
